// ContextOnlyModel_35270271435178
// MI455X (gfx1250) — matrix operand order repaired and checked by reading, not compiled, not run on hardware
//
#include <hip/hip_runtime.h>
#include <hip/hip_bf16.h>
#include <math.h>

// ---------------------------------------------------------------------------
// Shapes (fixed by the reference)
#define BB    16
#define TT    4096
#define VV    32000
#define DD    512
#define DD2   1024
#define SLOTS 256
#define ROWS  (BB*TT)          // 65536 tokens
#define MT    32               // tokens per workgroup in fused kernel
#define EPSV  1e-5f

typedef __attribute__((ext_vector_type(16))) __bf16 v16bf;
typedef __attribute__((ext_vector_type(8)))  __bf16 v8bf;
typedef __attribute__((ext_vector_type(8)))  float  v8f;

// ---------------------------------------------------------------------------
// Prep: transpose fp32 [K][N] -> bf16 [N][K]  (coalesced reads)
__global__ void k_transpose_bf16(const float* __restrict__ src,
                                 __bf16* __restrict__ dst,
                                 int K, int N, int total) {
  int idx = blockIdx.x * blockDim.x + threadIdx.x;
  if (idx >= total) return;
  int n = idx % N;
  int k = idx / N;
  dst[(long)n * K + k] = (__bf16)src[idx];
}

// ---------------------------------------------------------------------------
// Fused: embed-gather -> FFN (bf16 WMMA, f32 acc) -> residual -> LayerNorm -> gate
// One block = 32 tokens, 256 threads = 8 waves, 160 KB LDS (2 blocks/WGP).
// Each wave: GEMM1 = 2 N-passes x (2 M-frags x 4 N-tiles); GEMM2 = 2x4 frags.
// B-fragments are loaded once and feed both M-fragments (2x register reuse),
// halving L2 weight traffic vs a 16-token tile.
__global__ void __launch_bounds__(256)
k_fused_ffn_ln(const int*   __restrict__ seq,
               const float* __restrict__ emb,
               const __bf16* __restrict__ w1t,   // [1024][512] bf16
               const __bf16* __restrict__ w2t,   // [512][1024] bf16
               const float* __restrict__ b1,
               const float* __restrict__ b2,
               const float* __restrict__ ln_g,
               const float* __restrict__ ln_b,
               const float* __restrict__ wg,
               const float* __restrict__ bgp,
               float* __restrict__ hbuf,         // [ROWS][512] f32
               float* __restrict__ gate) {       // [ROWS] f32
  __shared__ float  sAf[MT * DD];    // h0 / x  fp32      (64 KB)
  __shared__ __bf16 sAb[MT * DD];    // h0 bf16 (GEMM1 A) (32 KB)
  __shared__ __bf16 sHb[MT * DD2];   // hidden bf16       (64 KB)

  const int tid = threadIdx.x;
  const int r0  = blockIdx.x * MT;

  // ---- stage 0: gather embedding rows for 32 tokens -----------------------
  {
    int m  = tid >> 3;             // token in tile (8 threads per row)
    int c0 = (tid & 7) * 64;       // 64 contiguous cols per thread
    int vid = seq[r0 + m];
    const float* erow = emb + (long)vid * DD;
    #pragma unroll
    for (int i = 0; i < 16; ++i) {
      float4 f4 = *(const float4*)(erow + c0 + i * 4);
      int c = c0 + i * 4;
      sAf[m * DD + c + 0] = f4.x;  sAb[m * DD + c + 0] = (__bf16)f4.x;
      sAf[m * DD + c + 1] = f4.y;  sAb[m * DD + c + 1] = (__bf16)f4.y;
      sAf[m * DD + c + 2] = f4.z;  sAb[m * DD + c + 2] = (__bf16)f4.z;
      sAf[m * DD + c + 3] = f4.w;  sAb[m * DD + c + 3] = (__bf16)f4.w;
    }
  }
  __syncthreads();

  const int wave = tid >> 5;
  const int lane = tid & 31;
  const int lrow = lane & 15;      // A: row M / B,D: col N
  const int lhi  = lane >> 4;      // half-select

  // ---- stage 1: hidden = relu(h0 @ w1 + b1) -------------------------------
  for (int npass = 0; npass < 2; ++npass) {
    v8f acc[2][4];
    #pragma unroll
    for (int mf = 0; mf < 2; ++mf)
      #pragma unroll
      for (int t = 0; t < 4; ++t)
        acc[mf][t] = (v8f){0.f,0.f,0.f,0.f,0.f,0.f,0.f,0.f};

    for (int kk = 0; kk < DD / 32; ++kk) {
      const int kb = kk * 32;
      v16bf a[2];
      #pragma unroll
      for (int mf = 0; mf < 2; ++mf) {
        const int ar = mf * 16 + lrow;
        v8bf a0 = *(const v8bf*)&sAb[ar * DD + kb + lhi * 16];
        v8bf a1 = *(const v8bf*)&sAb[ar * DD + kb + 8 + lhi * 16];
        a[mf] = __builtin_shufflevector(a0, a1, 0,1,2,3,4,5,6,7,8,9,10,11,12,13,14,15);
      }
      #pragma unroll
      for (int t = 0; t < 4; ++t) {
        const int n = (npass * 32 + wave * 4 + t) * 16 + lrow;
        const __bf16* bp = &w1t[(long)n * DD + kb + lhi * 16];
        if (kk + 1 < DD / 32) __builtin_prefetch(bp + 32, 0, 3);   // global_prefetch_b8
        v16bf b = *(const v16bf*)bp;
        acc[0][t] = __builtin_amdgcn_wmma_f32_16x16x32_bf16(
            false, a[0], false, b, (short)0, acc[0][t], false, false);
        acc[1][t] = __builtin_amdgcn_wmma_f32_16x16x32_bf16(
            false, a[1], false, b, (short)0, acc[1][t], false, false);
      }
    }
    #pragma unroll
    for (int t = 0; t < 4; ++t) {
      const int n = (npass * 32 + wave * 4 + t) * 16 + lrow;
      const float bias = b1[n];
      #pragma unroll
      for (int mf = 0; mf < 2; ++mf) {
        #pragma unroll
        for (int v = 0; v < 8; ++v) {
          const int m = mf * 16 + lhi * 8 + v;
          float hv = fmaxf(acc[mf][t][v] + bias, 0.f);
          sHb[m * DD2 + n] = (__bf16)hv;
        }
      }
    }
  }
  __syncthreads();

  // ---- stage 2: x = h0 + hidden @ w2 + b2 ---------------------------------
  {
    v8f acc[2][4];
    #pragma unroll
    for (int mf = 0; mf < 2; ++mf)
      #pragma unroll
      for (int t = 0; t < 4; ++t)
        acc[mf][t] = (v8f){0.f,0.f,0.f,0.f,0.f,0.f,0.f,0.f};

    for (int kk = 0; kk < DD2 / 32; ++kk) {
      const int kb = kk * 32;
      v16bf a[2];
      #pragma unroll
      for (int mf = 0; mf < 2; ++mf) {
        const int ar = mf * 16 + lrow;
        v8bf a0 = *(const v8bf*)&sHb[ar * DD2 + kb + lhi * 16];
        v8bf a1 = *(const v8bf*)&sHb[ar * DD2 + kb + 8 + lhi * 16];
        a[mf] = __builtin_shufflevector(a0, a1, 0,1,2,3,4,5,6,7,8,9,10,11,12,13,14,15);
      }
      #pragma unroll
      for (int t = 0; t < 4; ++t) {
        const int n = (wave * 4 + t) * 16 + lrow;
        const __bf16* bp = &w2t[(long)n * DD2 + kb + lhi * 16];
        if (kk + 1 < DD2 / 32) __builtin_prefetch(bp + 32, 0, 3);
        v16bf b = *(const v16bf*)bp;
        acc[0][t] = __builtin_amdgcn_wmma_f32_16x16x32_bf16(
            false, a[0], false, b, (short)0, acc[0][t], false, false);
        acc[1][t] = __builtin_amdgcn_wmma_f32_16x16x32_bf16(
            false, a[1], false, b, (short)0, acc[1][t], false, false);
      }
    }
    #pragma unroll
    for (int t = 0; t < 4; ++t) {
      const int n = (wave * 4 + t) * 16 + lrow;
      const float bias = b2[n];
      #pragma unroll
      for (int mf = 0; mf < 2; ++mf) {
        #pragma unroll
        for (int v = 0; v < 8; ++v) {
          const int m = mf * 16 + lhi * 8 + v;
          float xv = acc[mf][t][v] + bias + sAf[m * DD + n];  // residual
          sAf[m * DD + n] = xv;
        }
      }
    }
  }
  __syncthreads();

  // ---- stage 3: LayerNorm + gate; each wave owns 4 rows -------------------
  {
    const float bg0 = bgp[0];
    #pragma unroll
    for (int rr = 0; rr < 4; ++rr) {
      const int m = wave * 4 + rr;
      float s = 0.f, s2 = 0.f;
      #pragma unroll
      for (int i = 0; i < 16; ++i) {
        float x = sAf[m * DD + lane + i * 32];
        s += x; s2 += x * x;
      }
      #pragma unroll
      for (int o = 16; o > 0; o >>= 1) {
        s  += __shfl_xor(s,  o, 32);
        s2 += __shfl_xor(s2, o, 32);
      }
      const float mu  = s * (1.f / DD);
      const float var = s2 * (1.f / DD) - mu * mu;
      const float rs  = rsqrtf(var + EPSV);
      float gd = 0.f;
      const long row = (long)(r0 + m) * DD;
      #pragma unroll
      for (int i = 0; i < 16; ++i) {
        const int c = lane + i * 32;
        float hv = (sAf[m * DD + c] - mu) * rs * ln_g[c] + ln_b[c];
        hbuf[row + c] = hv;
        gd += hv * wg[c];
      }
      #pragma unroll
      for (int o = 16; o > 0; o >>= 1) gd += __shfl_xor(gd, o, 32);
      if (lane == 0) gate[r0 + m] = 1.f / (1.f + __expf(-(gd + bg0)));
    }
  }
}

// ---------------------------------------------------------------------------
// Top-256 of 4096 per batch (set only; order irrelevant — softmax+sum are
// permutation invariant).  Iterative argmax.
__global__ void k_topk(const float* __restrict__ gate, int* __restrict__ topidx) {
  const int b = blockIdx.x, tid = threadIdx.x;
  __shared__ float v[TT];
  __shared__ float smax[256];
  __shared__ int   sidx[256];
  for (int i = tid; i < TT; i += 256) v[i] = gate[b * TT + i];
  __syncthreads();
  for (int r = 0; r < SLOTS; ++r) {
    float bm = -INFINITY; int bi = 0;
    for (int i = tid; i < TT; i += 256) {
      float x = v[i];
      if (x > bm) { bm = x; bi = i; }
    }
    smax[tid] = bm; sidx[tid] = bi;
    __syncthreads();
    for (int s = 128; s > 0; s >>= 1) {
      if (tid < s && smax[tid + s] > smax[tid]) {
        smax[tid] = smax[tid + s]; sidx[tid] = sidx[tid + s];
      }
      __syncthreads();
    }
    if (tid == 0) { topidx[b * SLOTS + r] = sidx[0]; v[sidx[0]] = -INFINITY; }
    __syncthreads();
  }
}

// ---------------------------------------------------------------------------
// memory[b,s,:] = h[b, idx[b,s], :]
__global__ void k_gather(const float* __restrict__ h,
                         const int* __restrict__ topidx,
                         float* __restrict__ memv) {
  const int bs = blockIdx.x;           // b*256 + s
  const int b  = bs >> 8;
  const int t  = topidx[bs];
  const float* src = h + ((long)(b * TT + t)) * DD;
  float* dst = memv + (long)bs * DD;
  for (int c = threadIdx.x; c < DD; c += blockDim.x) dst[c] = src[c];
}

// ---------------------------------------------------------------------------
// q[b,:] = h[b,T-1,:] @ wq + bq   (tiny: plain FMA)
__global__ void k_qproj(const float* __restrict__ h,
                        const float* __restrict__ wq,
                        const float* __restrict__ bq,
                        float* __restrict__ q) {
  const int b = blockIdx.x, tid = threadIdx.x;
  __shared__ float qh[DD];
  const float* src = h + ((long)(b * TT + (TT - 1))) * DD;
  for (int c = tid; c < DD; c += 256) qh[c] = src[c];
  __syncthreads();
  for (int d = tid; d < DD; d += 256) {
    float acc = bq[d];
    for (int k = 0; k < DD; ++k) acc += qh[k] * wq[k * DD + d];
    q[b * DD + d] = acc;
  }
}

// ---------------------------------------------------------------------------
// scores = memory·q ; softmax ; ctx = attn·memory    (one block per batch)
__global__ void k_attn(const float* __restrict__ memv,
                       const float* __restrict__ q,
                       float* __restrict__ ctx) {
  const int b = blockIdx.x, tid = threadIdx.x;   // 256 threads
  __shared__ float sq[DD];
  __shared__ float sc[SLOTS];
  __shared__ float red[256];
  for (int c = tid; c < DD; c += 256) sq[c] = q[b * DD + c];
  __syncthreads();
  const float* mb = memv + (long)b * SLOTS * DD;
  const float* row = mb + tid * DD;
  float acc = 0.f;
  for (int k = 0; k < DD; ++k) acc += row[k] * sq[k];
  red[tid] = acc;
  __syncthreads();
  for (int s = 128; s > 0; s >>= 1) { if (tid < s) red[tid] = fmaxf(red[tid], red[tid + s]); __syncthreads(); }
  const float mx = red[0];
  __syncthreads();
  const float e = __expf(acc - mx);
  red[tid] = e;
  __syncthreads();
  for (int s = 128; s > 0; s >>= 1) { if (tid < s) red[tid] += red[tid + s]; __syncthreads(); }
  const float inv = 1.f / red[0];
  __syncthreads();
  sc[tid] = e * inv;
  __syncthreads();
  for (int d = tid; d < DD; d += 256) {
    float a = 0.f;
    for (int i = 0; i < SLOTS; ++i) a += sc[i] * mb[i * DD + d];
    ctx[b * DD + d] = a;
  }
}

// ---------------------------------------------------------------------------
// out = ctx @ wo + bo : (16 x 512) x (512 x 32000), bf16 WMMA.
// 250 blocks x 8 waves; each wave = one 16-col tile.
__global__ void __launch_bounds__(256)
k_out_gemm(const float* __restrict__ ctx,
           const float* __restrict__ bo,
           const __bf16* __restrict__ wot,    // [32000][512] bf16
           float* __restrict__ out) {
  __shared__ __bf16 sC[16 * DD];              // ctx bf16 (16 KB)
  const int tid = threadIdx.x;
  {
    int m  = tid >> 4;
    int c0 = (tid & 15) * 32;
    const float* crow = ctx + (long)m * DD;
    #pragma unroll
    for (int i = 0; i < 32; ++i) sC[m * DD + c0 + i] = (__bf16)crow[c0 + i];
  }
  __syncthreads();

  const int wave = tid >> 5;
  const int lane = tid & 31;
  const int lrow = lane & 15;
  const int lhi  = lane >> 4;
  const int n0   = (blockIdx.x * 8 + wave) * 16;

  v8f acc = (v8f){0.f,0.f,0.f,0.f,0.f,0.f,0.f,0.f};
  for (int kk = 0; kk < DD / 32; ++kk) {
    const int kb = kk * 32;
    v8bf a0 = *(const v8bf*)&sC[lrow * DD + kb + lhi * 16];
    v8bf a1 = *(const v8bf*)&sC[lrow * DD + kb + 8 + lhi * 16];
    v16bf a = __builtin_shufflevector(a0, a1, 0,1,2,3,4,5,6,7,8,9,10,11,12,13,14,15);
    const __bf16* bp = &wot[(long)(n0 + lrow) * DD + kb + lhi * 16];
    if (kk + 1 < DD / 32) __builtin_prefetch(bp + 32, 0, 3);
    v16bf b = *(const v16bf*)bp;
    acc = __builtin_amdgcn_wmma_f32_16x16x32_bf16(
        false, a, false, b, (short)0, acc, false, false);
  }
  const int n = n0 + lrow;
  const float bias = bo[n];
  #pragma unroll
  for (int v = 0; v < 8; ++v) {
    const int m = lhi * 8 + v;
    out[(long)m * VV + n] = acc[v] + bias;
  }
}

// ---------------------------------------------------------------------------
extern "C" void kernel_launch(void* const* d_in, const int* in_sizes, int n_in,
                              void* d_out, int out_size, void* d_ws, size_t ws_size,
                              hipStream_t stream) {
  const int*   seq  = (const int*)  d_in[0];
  const float* emb  = (const float*)d_in[1];
  const float* w1   = (const float*)d_in[2];
  const float* b1   = (const float*)d_in[3];
  const float* w2   = (const float*)d_in[4];
  const float* b2   = (const float*)d_in[5];
  const float* ln_g = (const float*)d_in[6];
  const float* ln_b = (const float*)d_in[7];
  const float* wg   = (const float*)d_in[8];
  const float* bg   = (const float*)d_in[9];
  const float* wq   = (const float*)d_in[10];
  const float* bq   = (const float*)d_in[11];
  const float* wo   = (const float*)d_in[12];
  const float* bo   = (const float*)d_in[13];

  char* ws = (char*)d_ws;
  size_t off = 0;
  auto carve = [&](size_t bytes) -> char* {
    char* p = ws + off;
    off += (bytes + 255) & ~(size_t)255;
    return p;
  };
  float*  hbuf   = (float*) carve((size_t)ROWS * DD * 4);      // 134 MB
  float*  gate   = (float*) carve((size_t)ROWS * 4);
  __bf16* w1t    = (__bf16*)carve((size_t)DD2 * DD * 2);
  __bf16* w2t    = (__bf16*)carve((size_t)DD * DD2 * 2);
  __bf16* wot    = (__bf16*)carve((size_t)VV * DD * 2);        // 32.8 MB
  int*    topidx = (int*)   carve((size_t)BB * SLOTS * 4);
  float*  memv   = (float*) carve((size_t)BB * SLOTS * DD * 4);
  float*  qv     = (float*) carve((size_t)BB * DD * 4);
  float*  ctxv   = (float*) carve((size_t)BB * DD * 4);

  // weight prep (bf16 transpose) — cached in 192 MB L2 for the GEMM kernels
  k_transpose_bf16<<<(DD * DD2 + 255) / 256, 256, 0, stream>>>(w1, w1t, DD,  DD2, DD * DD2);
  k_transpose_bf16<<<(DD2 * DD + 255) / 256, 256, 0, stream>>>(w2, w2t, DD2, DD,  DD2 * DD);
  k_transpose_bf16<<<(DD * VV + 255) / 256, 256, 0, stream>>>(wo, wot, DD,  VV,  DD * VV);

  // fused embed + FFN + residual + LN + gate  (32-token tiles)
  k_fused_ffn_ln<<<ROWS / MT, 256, 0, stream>>>(seq, emb, w1t, w2t, b1, b2,
                                                ln_g, ln_b, wg, bg, hbuf, gate);
  // selection + attention path
  k_topk  <<<BB,          256, 0, stream>>>(gate, topidx);
  k_gather<<<BB * SLOTS,  128, 0, stream>>>(hbuf, topidx, memv);
  k_qproj <<<BB,          256, 0, stream>>>(hbuf, wq, bq, qv);
  k_attn  <<<BB,          256, 0, stream>>>(memv, qv, ctxv);
  // vocab projection
  k_out_gemm<<<VV / 16 / 8, 256, 0, stream>>>(ctxv, bo, wot, (float*)d_out);
}
